// WindowAttention_67628555043000
// MI455X (gfx1250) — compile-verified, matrix operand order checked by reading, not run on hardware
//
#include <hip/hip_runtime.h>

typedef __bf16 bf16;
typedef __attribute__((ext_vector_type(16))) __bf16 v16bf;
typedef __attribute__((ext_vector_type(8)))  __bf16 v8bf;
typedef __attribute__((ext_vector_type(8)))  float  v8f;

// Problem constants
#define BATCH 1024
#define NQ    49     // query tokens per window
#define NK    196    // key tokens per window
#define CDIM  192
#define NHEAD 6
#define HDIM  32
#define MQP   64     // padded query rows (4 x 16)
#define NKP   208    // padded key rows   (13 x 16)
#define NKPP  224    // padded key rows for P*V K-loop (7 x 32)
#define QSCALE 0.17677669529663687f  // 32^-0.5

#define WMMA_BF16(a,b,c) \
  __builtin_amdgcn_wmma_f32_16x16x32_bf16(false,(a),false,(b),(short)0,(c),false,false)

// ---- fragment loaders (wave32, 16x16x32 layout) ---------------------------
// lane = hh*16 + r ; A: r = M-row, B: r = N-col.
// VGPR v holds K = 16*(v/4) + 8*hh + 2*(v%4)  => two contiguous 8-elt runs:
//   run0 at k_base + 8*hh, run1 at k_base + 16 + 8*hh.
// Caller passes p = row_ptr + k_base + 8*hh (must be 16B aligned).
__device__ __forceinline__ v16bf frag_load_bf(const bf16* p) {
  v8bf lo = *(const v8bf*)p;
  v8bf hi = *(const v8bf*)(p + 16);
  v16bf f;
#pragma unroll
  for (int i = 0; i < 8; ++i) { f[i] = lo[i]; f[i + 8] = hi[i]; }
  return f;
}
__device__ __forceinline__ v16bf frag_load_f32(const float* p) {
  v8f lo = *(const v8f*)p;
  v8f hi = *(const v8f*)(p + 16);
  v16bf f;
#pragma unroll
  for (int i = 0; i < 8; ++i) { f[i] = (bf16)lo[i]; f[i + 8] = (bf16)hi[i]; }
  return f;
}
__device__ __forceinline__ v16bf frag_zero() {
  v16bf f;
#pragma unroll
  for (int i = 0; i < 16; ++i) f[i] = (bf16)0.0f;
  return f;
}

// ---- kernel 0: transpose + f32->bf16 convert weight matrix ----------------
// W is (K x N) row-major (in,out); Wt becomes (N x K) row-major bf16.
__global__ void wa_convT(const float* __restrict__ W, bf16* __restrict__ Wt,
                         int K, int N) {
  int idx = blockIdx.x * 256 + threadIdx.x;
  if (idx < K * N) {
    int k = idx / N, n = idx % N;
    Wt[(size_t)n * K + k] = (bf16)W[idx];
  }
}

// ---- kernel 1: q = scale * (x1 @ Wqkv + b), bf16 [B,64,192] ---------------
// Register blocked: wave owns (mtile, 6 ntiles); A loaded once per K-step.
__global__ void wa_qproj(const float* __restrict__ x1, const bf16* __restrict__ Wt,
                         const float* __restrict__ bias, bf16* __restrict__ qout) {
  const int b    = blockIdx.x;
  const int lane = threadIdx.x & 31, wave = threadIdx.x >> 5;
  const int hh   = lane >> 4, r = lane & 15;
  const int mt   = wave & 3;          // 4 M-tiles
  const int ng   = (wave >> 2) * 6;   // ntile group base: 0 or 6
  const float* xb = x1 + (size_t)b * NQ * CDIM;
  bf16* qb = qout + (size_t)b * MQP * CDIM;
  const int m = mt * 16 + r;

  v8f acc[6];
#pragma unroll
  for (int j = 0; j < 6; ++j) {
    const float cb = bias[(ng + j) * 16 + r];
#pragma unroll
    for (int i = 0; i < 8; ++i) acc[j][i] = cb;
  }
#pragma unroll
  for (int ks = 0; ks < 6; ++ks) {
    v16bf a = (m < NQ) ? frag_load_f32(xb + m * CDIM + ks * 32 + hh * 8)
                       : frag_zero();
#pragma unroll
    for (int j = 0; j < 6; ++j) {
      v16bf w = frag_load_bf(Wt + ((ng + j) * 16 + r) * CDIM + ks * 32 + hh * 8);
      acc[j] = WMMA_BF16(a, w, acc[j]);
    }
  }
#pragma unroll
  for (int j = 0; j < 6; ++j) {
    const int n = (ng + j) * 16 + r;
#pragma unroll
    for (int v = 0; v < 8; ++v) {
      int row = mt * 16 + v + 8 * hh;
      qb[row * CDIM + n] = (bf16)(acc[j][v] * QSCALE);
    }
  }
}

// ---- kernel 2: kv = x2 @ Wkv + b ; k bf16 [B,208,192], v^T bf16 [B,192,224]
// Wave owns 3 ntiles, loops 13 mtiles; A loaded once per (mt, K-step).
__global__ void wa_kvproj(const float* __restrict__ x2, const bf16* __restrict__ Wt,
                          const float* __restrict__ bias,
                          bf16* __restrict__ kout, bf16* __restrict__ vtout) {
  const int b    = blockIdx.x;
  const int lane = threadIdx.x & 31, wave = threadIdx.x >> 5;
  const int hh   = lane >> 4, r = lane & 15;
  const int nb   = wave * 3;                     // 8 waves x 3 ntiles = 24
  const float* xb = x2 + (size_t)b * NK * CDIM;
  bf16* kb  = kout  + (size_t)b * NKP * CDIM;
  bf16* vtb = vtout + (size_t)b * CDIM * NKPP;
  // zero v^T padding rows m in [208,224) so 0 * v_pad == 0 in P*V
  for (int i = threadIdx.x; i < CDIM * 16; i += 256) {
    int d = i >> 4, mm = 208 + (i & 15);
    vtb[d * NKPP + mm] = (bf16)0.0f;
  }
  for (int mt = 0; mt < 13; ++mt) {
    const int m = mt * 16 + r;                   // [0,208)
    v8f acc[3];
#pragma unroll
    for (int j = 0; j < 3; ++j) {
      const float cb = bias[(nb + j) * 16 + r];
#pragma unroll
      for (int i = 0; i < 8; ++i) acc[j][i] = cb;
    }
#pragma unroll
    for (int ks = 0; ks < 6; ++ks) {
      v16bf a = (m < NK) ? frag_load_f32(xb + m * CDIM + ks * 32 + hh * 8)
                         : frag_zero();
#pragma unroll
      for (int j = 0; j < 3; ++j) {
        v16bf w = frag_load_bf(Wt + ((nb + j) * 16 + r) * CDIM + ks * 32 + hh * 8);
        acc[j] = WMMA_BF16(a, w, acc[j]);
      }
    }
#pragma unroll
    for (int j = 0; j < 3; ++j) {
      const int n = (nb + j) * 16 + r;           // [0,384)
#pragma unroll
      for (int v = 0; v < 8; ++v) {
        int row = mt * 16 + v + 8 * hh;          // < 208
        if (n < CDIM) kb[row * CDIM + n] = (bf16)acc[j][v];
        else          vtb[(n - CDIM) * NKPP + row] = (bf16)acc[j][v];
      }
    }
  }
}

// ---- kernel 3: per (window, head, half-M): S=q k^T, +bias, softmax, O=P v -
__global__ void wa_attn(const bf16* __restrict__ qbuf, const bf16* __restrict__ kbuf,
                        const bf16* __restrict__ vtbuf, const float* __restrict__ table,
                        bf16* __restrict__ obuf) {
  __shared__ float S[32 * NKP];        // 32 x 208 f32 scores
  __shared__ bf16  P[32 * NKPP];       // 32 x 224 bf16 probabilities
  __shared__ float biasH[400];
  const int id = blockIdx.x;
  const int b = id / 12, rem = id % 12, h = rem >> 1, hb = rem & 1;
  const int mbase = hb * 32;
  const int lane = threadIdx.x & 31, wave = threadIdx.x >> 5;
  const int hh = lane >> 4, r = lane & 15;

  // stage this head's bias column via async gather to LDS (ASYNCcnt path)
  for (int i = threadIdx.x; i < 400; i += 256) {
    const float* gp = table + (size_t)i * NHEAD + h;
    unsigned ldsoff = (unsigned)(unsigned long long)(biasH + i);
    asm volatile("global_load_async_to_lds_b32 %0, %1, off"
                 :: "v"(ldsoff), "v"(gp) : "memory");
  }
  for (int i = threadIdx.x; i < 32 * NKPP; i += 256) P[i] = (bf16)0.0f;
  asm volatile("s_wait_asynccnt 0x0" ::: "memory");
  __syncthreads();

  // S = q * k^T  (2 Mtiles x 13 Ntiles, K = 32 = one WMMA step)
  const bf16* qb = qbuf + ((size_t)b * MQP + mbase) * CDIM + h * HDIM;
  const bf16* kb = kbuf + (size_t)b * NKP * CDIM + h * HDIM;
  for (int t = wave; t < 26; t += 8) {
    const int mt = t / 13, nt = t % 13;
    v16bf a = frag_load_bf(qb + (mt * 16 + r) * CDIM + hh * 8);
    v16bf k = frag_load_bf(kb + (nt * 16 + r) * CDIM + hh * 8);
    v8f c = {};
    c = WMMA_BF16(a, k, c);
#pragma unroll
    for (int v = 0; v < 8; ++v)
      S[(mt * 16 + v + 8 * hh) * NKP + nt * 16 + r] = c[v];
  }
  __syncthreads();

  // wave-parallel softmax: each wave owns 4 rows; lanes stride the columns
  for (int rr = 0; rr < 4; ++rr) {
    const int row = wave * 4 + rr;               // 0..31 local
    const int g = mbase + row;                   // global query row
    if (g < NQ) {
      const int r1 = g / 7, c1 = g % 7;
      float vals[7];
      float mx = -1e30f;
#pragma unroll
      for (int it = 0; it < 7; ++it) {
        const int m = it * 32 + lane;
        float v = -1e30f;
        if (m < NK) {
          int r2 = m / 14, c2 = m % 14;
          v = S[row * NKP + m] + biasH[(r1 - r2 + 13) * 20 + (c1 - c2 + 13)];
        }
        vals[it] = v;
        mx = fmaxf(mx, v);
      }
#pragma unroll
      for (int off = 16; off > 0; off >>= 1)
        mx = fmaxf(mx, __shfl_xor(mx, off, 32));
      float sum = 0.0f;
#pragma unroll
      for (int it = 0; it < 7; ++it) {
        const int m = it * 32 + lane;
        float e = 0.0f;
        if (m < NK) e = __expf(vals[it] - mx);
        vals[it] = e;
        sum += e;
      }
#pragma unroll
      for (int off = 16; off > 0; off >>= 1)
        sum += __shfl_xor(sum, off, 32);
      const float inv = 1.0f / sum;
#pragma unroll
      for (int it = 0; it < 7; ++it) {
        const int m = it * 32 + lane;
        if (m < NK) P[row * NKPP + m] = (bf16)(vals[it] * inv);
      }
    }
  }
  __syncthreads();

  // O = P * v  (2 Mtiles x 2 Ntiles, K = 224 = 7 WMMA steps)
  const bf16* vtb = vtbuf + ((size_t)b * CDIM + h * HDIM) * NKPP;
  bf16* ob = obuf + (size_t)b * MQP * CDIM + h * HDIM;
  if (wave < 4) {
    const int mt = wave >> 1, nt = wave & 1;
    v8f c = {};
#pragma unroll
    for (int ks = 0; ks < 7; ++ks) {
      v16bf a = frag_load_bf(P + (mt * 16 + r) * NKPP + ks * 32 + hh * 8);
      v16bf v = frag_load_bf(vtb + (nt * 16 + r) * NKPP + ks * 32 + hh * 8);
      c = WMMA_BF16(a, v, c);
    }
#pragma unroll
    for (int v = 0; v < 8; ++v) {
      int row = mbase + mt * 16 + v + 8 * hh;
      ob[row * CDIM + nt * 16 + r] = (bf16)c[v];
    }
  }
}

// ---- kernel 4: out = O @ Wproj + b, f32 [B,49,192] ------------------------
__global__ void wa_oproj(const bf16* __restrict__ obuf, const bf16* __restrict__ Wt,
                         const float* __restrict__ bias, float* __restrict__ out) {
  const int b    = blockIdx.x;
  const int lane = threadIdx.x & 31, wave = threadIdx.x >> 5;
  const int hh   = lane >> 4, r = lane & 15;
  const int mt   = wave & 3;
  const int ng   = (wave >> 2) * 6;
  const bf16* ob = obuf + (size_t)b * MQP * CDIM;
  float* yb = out + (size_t)b * NQ * CDIM;

  v8f acc[6];
#pragma unroll
  for (int j = 0; j < 6; ++j) {
    const float cb = bias[(ng + j) * 16 + r];
#pragma unroll
    for (int i = 0; i < 8; ++i) acc[j][i] = cb;
  }
#pragma unroll
  for (int ks = 0; ks < 6; ++ks) {
    v16bf a = frag_load_bf(ob + (mt * 16 + r) * CDIM + ks * 32 + hh * 8);
#pragma unroll
    for (int j = 0; j < 6; ++j) {
      v16bf w = frag_load_bf(Wt + ((ng + j) * 16 + r) * CDIM + ks * 32 + hh * 8);
      acc[j] = WMMA_BF16(a, w, acc[j]);
    }
  }
#pragma unroll
  for (int j = 0; j < 6; ++j) {
    const int n = (ng + j) * 16 + r;
#pragma unroll
    for (int v = 0; v < 8; ++v) {
      int row = mt * 16 + v + 8 * hh;
      if (row < NQ) yb[row * CDIM + n] = acc[j][v];
    }
  }
}

extern "C" void kernel_launch(void* const* d_in, const int* in_sizes, int n_in,
                              void* d_out, int out_size, void* d_ws, size_t ws_size,
                              hipStream_t stream) {
  const float* x1      = (const float*)d_in[0];
  const float* x2      = (const float*)d_in[1];
  const float* qkv_w   = (const float*)d_in[2];
  const float* qkv_b   = (const float*)d_in[3];
  const float* qkv2_w  = (const float*)d_in[4];
  const float* qkv2_b  = (const float*)d_in[5];
  const float* proj_w  = (const float*)d_in[6];
  const float* proj_b  = (const float*)d_in[7];
  const float* table   = (const float*)d_in[8];
  float* out = (float*)d_out;

  char* ws = (char*)d_ws;
  size_t off = 0;
  bf16* wt_qkv  = (bf16*)(ws + off); off += (size_t)CDIM * CDIM * 2;        //  72 KB
  bf16* wt_kv   = (bf16*)(ws + off); off += (size_t)(2 * CDIM) * CDIM * 2;  // 144 KB
  bf16* wt_proj = (bf16*)(ws + off); off += (size_t)CDIM * CDIM * 2;        //  72 KB
  bf16* qbuf    = (bf16*)(ws + off); off += (size_t)BATCH * MQP * CDIM * 2; //  24 MB
  bf16* kbuf    = (bf16*)(ws + off); off += (size_t)BATCH * NKP * CDIM * 2; //  78 MB
  bf16* vtbuf   = (bf16*)(ws + off); off += (size_t)BATCH * CDIM * NKPP * 2;//  84 MB
  bf16* obuf    = (bf16*)(ws + off); off += (size_t)BATCH * MQP * CDIM * 2; //  24 MB
  (void)ws_size; (void)in_sizes; (void)n_in; (void)out_size;

  wa_convT<<<(CDIM * CDIM + 255) / 256, 256, 0, stream>>>(qkv_w,  wt_qkv,  CDIM, CDIM);
  wa_convT<<<(CDIM * 2 * CDIM + 255) / 256, 256, 0, stream>>>(qkv2_w, wt_kv, CDIM, 2 * CDIM);
  wa_convT<<<(CDIM * CDIM + 255) / 256, 256, 0, stream>>>(proj_w, wt_proj, CDIM, CDIM);

  wa_qproj <<<BATCH, 256, 0, stream>>>(x1, wt_qkv, qkv_b, qbuf);
  wa_kvproj<<<BATCH, 256, 0, stream>>>(x2, wt_kv, qkv2_b, kbuf, vtbuf);
  wa_attn  <<<BATCH * NHEAD * 2, 256, 0, stream>>>(qbuf, kbuf, vtbuf, table, obuf);
  wa_oproj <<<BATCH, 256, 0, stream>>>(obuf, wt_proj, proj_b, out);
}
